// QuadrilateralDetection_30769145709154
// MI455X (gfx1250) — matrix operand order repaired and checked by reading, not compiled, not run on hardware
//
#include <hip/hip_runtime.h>
#include <hip/hip_bf16.h>

// ---------------------------------------------------------------------------
// MI455X (gfx1250): 256x256-K GEMMs on v_wmma_f32_16x16x32_f16 (f16 in, f32 acc).
// A-activations staged block-wide into LDS via GLOBAL_LOAD_ASYNC_TO_LDS_B128
// (ASYNCcnt / s_wait_asynccnt), B-weights streamed from global (L2-resident).
// ---------------------------------------------------------------------------

typedef __attribute__((ext_vector_type(16))) _Float16 v16h;
typedef __attribute__((ext_vector_type(8)))  _Float16 v8h;
typedef __attribute__((ext_vector_type(8)))  float    v8f;

#define CDIM  256
#define LDP   272          // padded LDS row stride (floats)
#define NPOS  5376         // 64*64 + 32*32 + 16*16
#define BATCH 16
#define MAXI  100

__device__ __forceinline__ float silu_f(float x)    { return x / (1.f + __expf(-x)); }
__device__ __forceinline__ float sigmoid_f(float x) { return 1.f / (1.f + __expf(-x)); }

// ---- gfx1250 async global->LDS copy (ASYNCcnt) ----
// dsaddr VGPR takes the low 32 bits of the generic shared-memory address
// (flat aperture rule).
__device__ __forceinline__ void async_copy_b128(unsigned lds_off, unsigned long long gaddr) {
  asm volatile("global_load_async_to_lds_b128 %0, %1, off"
               :: "v"(lds_off), "v"(gaddr) : "memory");
}
__device__ __forceinline__ void wait_async_zero() {
  asm volatile("s_wait_asynccnt 0" ::: "memory");
}

// ---- WMMA fragment loads, contiguous k order shared by A and B ----
// A 16x32 (MxK) f16: lanes 0-15 = rows M, K 0..15; lanes 16-31 = rows M, K 16..31
__device__ __forceinline__ v16h load_a_frag(const _Float16* __restrict__ Xr, int k0, int lane) {
  int half = lane >> 4;
  int m    = lane & 15;
  const _Float16* p = Xr + (size_t)m * CDIM + k0 + half * 16;
  return *(const v16h*)(p);
}
// B 32x16 (KxN) f16 from W^T stored row-major [N][K]:
// lanes 0-15 = col N, K 0..15; lanes 16-31 = col N, K 16..31 (one 32B load)
__device__ __forceinline__ v16h load_b_frag(const _Float16* __restrict__ Wt, int n0, int k0, int lane) {
  int half = lane >> 4;
  int n    = lane & 15;
  return *(const v16h*)(Wt + (size_t)(n0 + n) * CDIM + k0 + half * 16);
}
__device__ __forceinline__ v8f wmma16(v16h a, v16h b, v8f c) {
  return __builtin_amdgcn_wmma_f32_16x16x32_f16(false, a, false, b, (short)0, c, false, false);
}

// ---------------------------------------------------------------------------
// Weight transpose + f16 convert: src f32 [nmat][K=256][N=256] -> dst f16 [nmat][N][K]
// ---------------------------------------------------------------------------
__global__ void __launch_bounds__(256) k_transpose_w(const float* __restrict__ src,
                                                     _Float16* __restrict__ dst) {
  __shared__ float t[32][33];
  int m  = blockIdx.z;
  const float* S = src + (size_t)m * 65536;
  _Float16*    D = dst + (size_t)m * 65536;
  int n0 = blockIdx.x * 32, k0 = blockIdx.y * 32;
  int li = threadIdx.x & 31, lj = threadIdx.x >> 5;
#pragma unroll
  for (int r = 0; r < 32; r += 8)
    t[lj + r][li] = S[(size_t)(k0 + lj + r) * CDIM + n0 + li];
  __syncthreads();
#pragma unroll
  for (int r = 0; r < 32; r += 8)
    D[(size_t)(n0 + lj + r) * CDIM + k0 + li] = (_Float16)t[li][lj + r];
}

// ---------------------------------------------------------------------------
// Input transpose: x f32 [B][C][HW] -> A f16 [B][HW][C]
// ---------------------------------------------------------------------------
__global__ void __launch_bounds__(256) k_transpose_in(const float* __restrict__ src,
                                                      _Float16* __restrict__ dst, int HW) {
  __shared__ float t[32][33];
  int b   = blockIdx.z;
  int hw0 = blockIdx.x * 32, c0 = blockIdx.y * 32;
  const float* S = src + (size_t)b * CDIM * HW;
  _Float16*    D = dst + (size_t)b * HW * CDIM;
  int li = threadIdx.x & 31, lj = threadIdx.x >> 5;
#pragma unroll
  for (int r = 0; r < 32; r += 8)
    t[lj + r][li] = S[(size_t)(c0 + lj + r) * HW + hw0 + li];
  __syncthreads();
#pragma unroll
  for (int r = 0; r < 32; r += 8)
    D[(size_t)(hw0 + lj + r) * CDIM + c0 + li] = (_Float16)t[li][lj + r];
}

// ---------------------------------------------------------------------------
// Global context: gc[b,n] = mean_p silu(conv(x5)[p,n]*s[n]+b[n]); grid (16 ntiles, 16 b)
// Each wave owns distinct rows -> no cross-wave A redundancy, keep global loads.
// ---------------------------------------------------------------------------
__global__ void __launch_bounds__(256) k_gc(const _Float16* __restrict__ A5,
                                            const _Float16* __restrict__ Wt,
                                            const float* __restrict__ sc,
                                            const float* __restrict__ bi,
                                            float* __restrict__ gc) {
  __shared__ float colsum[16];
  int tid = threadIdx.x, lane = tid & 31, wave = tid >> 5;
  int b = blockIdx.y, n0 = blockIdx.x * 16;
  if (tid < 16) colsum[tid] = 0.f;
  __syncthreads();
  const _Float16* Ab = A5 + (size_t)b * 256 * CDIM;
  const _Float16* X0 = Ab + (size_t)(wave * 16) * CDIM;        // rows wave*16..+15
  const _Float16* X1 = Ab + (size_t)((wave + 8) * 16) * CDIM;  // rows (wave+8)*16..+15
  v8f c0 = {}, c1 = {};
#pragma unroll
  for (int k0 = 0; k0 < CDIM; k0 += 32) {
    v16h a0 = load_a_frag(X0, k0, lane);
    v16h a1 = load_a_frag(X1, k0, lane);
    v16h bm = load_b_frag(Wt, n0, k0, lane);
    c0 = wmma16(a0, bm, c0);
    c1 = wmma16(a1, bm, c1);
  }
  int cl = lane & 15;
  float s = sc[n0 + cl], bb = bi[n0 + cl];
  float part = 0.f;
#pragma unroll
  for (int v = 0; v < 8; ++v) {
    part += silu_f(c0[v] * s + bb);
    part += silu_f(c1[v] * s + bb);
  }
  atomicAdd(&colsum[cl], part);
  __syncthreads();
  if (tid < 16) gc[b * CDIM + n0 + tid] = colsum[tid] * (1.f / 256.f);
}

// ---------------------------------------------------------------------------
// Lateral conv: feats = silu(A@W * s + b) + gc[b].  One block = 16 rows x all 256 cols.
// A block (8KB) async-staged into LDS, shared by all 8 waves.
// ---------------------------------------------------------------------------
__global__ void __launch_bounds__(256) k_conv(const _Float16* __restrict__ A,
                                              _Float16* __restrict__ feats,
                                              const _Float16* __restrict__ Wt,
                                              const float* __restrict__ sc,
                                              const float* __restrict__ bi,
                                              const float* __restrict__ gc,
                                              int HW, int lvl_off) {
  __shared__ _Float16 smA[16 * CDIM];   // 8 KB A block
  __shared__ float tile[16 * LDP];
  int tid = threadIdx.x, lane = tid & 31, wave = tid >> 5;
  size_t m0 = (size_t)blockIdx.x * 16;
  int b = (int)(m0 / (size_t)HW);
  int local = (int)(m0 % (size_t)HW);

  // async global -> LDS: 16 rows x 256 f16 = 8192 B = 256 threads x 2 x 16 B
  {
    unsigned           l0 = (unsigned)(size_t)(&smA[0]) + (unsigned)tid * 16u;
    unsigned long long g0 = (unsigned long long)(size_t)(A + m0 * CDIM) +
                            (unsigned long long)tid * 16u;
    async_copy_b128(l0, g0);
    async_copy_b128(l0 + 4096u, g0 + 4096ull);
    wait_async_zero();
  }
  __syncthreads();

  int n0a = wave * 32, n0b = wave * 32 + 16;
  v8f c0 = {}, c1 = {};
#pragma unroll
  for (int k0 = 0; k0 < CDIM; k0 += 32) {
    v16h a  = load_a_frag(smA, k0, lane);
    v16h b0 = load_b_frag(Wt, n0a, k0, lane);
    v16h b1 = load_b_frag(Wt, n0b, k0, lane);
    c0 = wmma16(a, b0, c0);
    c1 = wmma16(a, b1, c1);
  }
  int half = lane >> 4, cl = lane & 15;
#pragma unroll
  for (int v = 0; v < 8; ++v) {
    int r = v + half * 8;
    tile[r * LDP + n0a + cl] = c0[v];
    tile[r * LDP + n0b + cl] = c1[v];
  }
  __syncthreads();
  int r = tid >> 4, cb = (tid & 15) * 16;
  const float* gcb = gc + b * CDIM;
  _Float16 outv[16];
#pragma unroll
  for (int i = 0; i < 16; ++i) {
    int c = cb + i;
    float x = tile[r * LDP + c] * sc[c] + bi[c];
    outv[i] = (_Float16)(silu_f(x) + gcb[c]);
  }
  _Float16* Yp = feats + ((size_t)b * NPOS + lvl_off + local + r) * CDIM + cb;
  *(v8h*)Yp       = *(v8h*)&outv[0];
  *(v8h*)(Yp + 8) = *(v8h*)&outv[8];
}

// ---------------------------------------------------------------------------
// MLP hidden layer: Y = silu(LN(X@W + hb) * g + be). One block = 16 rows x 256 cols.
// A block async-staged into LDS; LN reduction via LDS.
// ---------------------------------------------------------------------------
__global__ void __launch_bounds__(256) k_mlp_hidden(const _Float16* __restrict__ X,
                                                    _Float16* __restrict__ Y,
                                                    const _Float16* __restrict__ Wt,
                                                    const float* __restrict__ hb,
                                                    const float* __restrict__ g,
                                                    const float* __restrict__ be) {
  __shared__ _Float16 smA[16 * CDIM];   // 8 KB A block
  __shared__ float tile[16 * LDP];
  __shared__ float red_s[256], red_q[256];
  __shared__ float mrow[16], rrow[16];
  int tid = threadIdx.x, lane = tid & 31, wave = tid >> 5;
  size_t m0 = (size_t)blockIdx.x * 16;

  {
    unsigned           l0 = (unsigned)(size_t)(&smA[0]) + (unsigned)tid * 16u;
    unsigned long long g0 = (unsigned long long)(size_t)(X + m0 * CDIM) +
                            (unsigned long long)tid * 16u;
    async_copy_b128(l0, g0);
    async_copy_b128(l0 + 4096u, g0 + 4096ull);
    wait_async_zero();
  }
  __syncthreads();

  int n0a = wave * 32, n0b = wave * 32 + 16;
  v8f c0 = {}, c1 = {};
#pragma unroll
  for (int k0 = 0; k0 < CDIM; k0 += 32) {
    v16h a  = load_a_frag(smA, k0, lane);
    v16h b0 = load_b_frag(Wt, n0a, k0, lane);
    v16h b1 = load_b_frag(Wt, n0b, k0, lane);
    c0 = wmma16(a, b0, c0);
    c1 = wmma16(a, b1, c1);
  }
  int half = lane >> 4, cl = lane & 15;
  float bb0 = hb[n0a + cl], bb1 = hb[n0b + cl];
#pragma unroll
  for (int v = 0; v < 8; ++v) {
    int r = v + half * 8;
    tile[r * LDP + n0a + cl] = c0[v] + bb0;
    tile[r * LDP + n0b + cl] = c1[v] + bb1;
  }
  __syncthreads();
  int r = tid >> 4, cb = (tid & 15) * 16;
  float s = 0.f, q = 0.f;
#pragma unroll
  for (int i = 0; i < 16; ++i) {
    float x = tile[r * LDP + cb + i];
    s += x; q += x * x;
  }
  red_s[tid] = s; red_q[tid] = q;
  __syncthreads();
  if (tid < 16) {
    float S = 0.f, Q = 0.f;
#pragma unroll
    for (int j = 0; j < 16; ++j) { S += red_s[tid * 16 + j]; Q += red_q[tid * 16 + j]; }
    float mean = S * (1.f / CDIM);
    float var  = Q * (1.f / CDIM) - mean * mean;
    mrow[tid] = mean;
    rrow[tid] = rsqrtf(var + 1e-5f);
  }
  __syncthreads();
  float mean = mrow[r], rstd = rrow[r];
  _Float16 outv[16];
#pragma unroll
  for (int i = 0; i < 16; ++i) {
    int c = cb + i;
    float x = (tile[r * LDP + c] - mean) * rstd * g[c] + be[c];
    outv[i] = (_Float16)silu_f(x);
  }
  _Float16* Yp = Y + (m0 + r) * CDIM + cb;
  *(v8h*)Yp       = *(v8h*)&outv[0];
  *(v8h*)(Yp + 8) = *(v8h*)&outv[8];
}

// ---------------------------------------------------------------------------
// loc final: logits[row] = act[row,:] . w + ob.  One wave per row, 8 rows/block.
// ---------------------------------------------------------------------------
__global__ void __launch_bounds__(256) k_loc_final(const _Float16* __restrict__ X,
                                                   const float* __restrict__ w,
                                                   const float* __restrict__ ob,
                                                   float* __restrict__ logits) {
  int lane = threadIdx.x & 31, wave = threadIdx.x >> 5;
  size_t row = (size_t)blockIdx.x * 8 + wave;
  const _Float16* xr = X + row * CDIM + lane * 8;
  const float* wp = w + lane * 8;
  float p = 0.f;
#pragma unroll
  for (int i = 0; i < 8; ++i) p += (float)xr[i] * wp[i];
  for (int o = 16; o > 0; o >>= 1) p += __shfl_down(p, o, 32);
  if (lane == 0) logits[row] = p + ob[0];
}

// ---------------------------------------------------------------------------
// top-k (k=100) per image; writes scores + num_instances into d_out, indices to ws.
// ---------------------------------------------------------------------------
__global__ void __launch_bounds__(256) k_topk(const float* __restrict__ logits,
                                              int* __restrict__ top_idx,
                                              float* __restrict__ out) {
  __shared__ float vals[NPOS];
  __shared__ float rv[256];
  __shared__ int   ri[256];
  __shared__ float sv[MAXI];
  __shared__ int   si[MAXI];
  int b = blockIdx.x, tid = threadIdx.x;
  const float* L = logits + (size_t)b * NPOS;
  for (int i = tid; i < NPOS; i += 256) vals[i] = L[i];
  __syncthreads();
  for (int it = 0; it < MAXI; ++it) {
    float bv = -3.0e38f; int bi = 0;
    for (int i = tid; i < NPOS; i += 256)
      if (vals[i] > bv) { bv = vals[i]; bi = i; }
    rv[tid] = bv; ri[tid] = bi;
    __syncthreads();
    for (int s = 128; s > 0; s >>= 1) {
      if (tid < s) {
        if (rv[tid + s] > rv[tid] ||
            (rv[tid + s] == rv[tid] && ri[tid + s] < ri[tid])) {
          rv[tid] = rv[tid + s]; ri[tid] = ri[tid + s];
        }
      }
      __syncthreads();
    }
    if (tid == 0) { sv[it] = rv[0]; si[it] = ri[0]; vals[ri[0]] = -3.0e38f; }
    __syncthreads();
  }
  if (tid < MAXI) {
    out[16 + b * MAXI + tid] = sigmoid_f(sv[tid]);
    top_idx[b * MAXI + tid]  = si[tid];
  }
  __syncthreads();
  if (tid == 0) {
    int cnt = 0;
    for (int i = 0; i < MAXI; ++i) cnt += (sigmoid_f(sv[i]) > 0.5f) ? 1 : 0;
    out[b] = (float)cnt;
  }
}

// ---------------------------------------------------------------------------
// Gather selected feature rows (f16, 512B each), 128 dwords per row.
// ---------------------------------------------------------------------------
__global__ void __launch_bounds__(128) k_gather(const _Float16* __restrict__ feats,
                                                const int* __restrict__ top_idx,
                                                _Float16* __restrict__ sel) {
  int row = blockIdx.x;
  int b   = row / MAXI;
  int n   = top_idx[row];
  const unsigned* s = (const unsigned*)(feats + ((size_t)b * NPOS + n) * CDIM);
  unsigned*       d = (unsigned*)(sel + (size_t)row * CDIM);
  d[threadIdx.x] = s[threadIdx.x];
}

// ---------------------------------------------------------------------------
// cls final: argmax over 80 logits per row. One wave per row, 8 rows/block.
// ---------------------------------------------------------------------------
__global__ void __launch_bounds__(256) k_cls_final(const _Float16* __restrict__ X,
                                                   const float* __restrict__ W,
                                                   const float* __restrict__ ob,
                                                   float* __restrict__ out) {
  __shared__ float rb[8][CDIM];
  int tid = threadIdx.x, lane = tid & 31, wave = tid >> 5;
  int row = blockIdx.x * 8 + wave;
  const _Float16* xr = X + (size_t)row * CDIM;
  for (int i = lane; i < CDIM; i += 32) rb[wave][i] = (float)xr[i];
  __syncthreads();
  float best = -3.0e38f; int bc = 0;
  for (int c = lane; c < 80; c += 32) {
    float acc = ob[c];
    for (int k = 0; k < CDIM; ++k) acc += rb[wave][k] * W[k * 80 + c];
    if (acc > best) { best = acc; bc = c; }
  }
  for (int o = 16; o > 0; o >>= 1) {
    float ov = __shfl_down(best, o, 32);
    int   oc = __shfl_down(bc, o, 32);
    if (ov > best || (ov == best && oc < bc)) { best = ov; bc = oc; }
  }
  if (lane == 0) out[row] = (float)bc;
}

// ---------------------------------------------------------------------------
// quad final: out = (grid_offset + tanh(act@W + ob)) * 512. One wave per row.
// ---------------------------------------------------------------------------
__global__ void __launch_bounds__(256) k_quad_final(const _Float16* __restrict__ X,
                                                    const float* __restrict__ W,
                                                    const float* __restrict__ ob,
                                                    const int* __restrict__ top_idx,
                                                    float* __restrict__ out) {
  __shared__ float rb[8][CDIM];
  int tid = threadIdx.x, lane = tid & 31, wave = tid >> 5;
  int row = blockIdx.x * 8 + wave;
  const _Float16* xr = X + (size_t)row * CDIM;
  for (int i = lane; i < CDIM; i += 32) rb[wave][i] = (float)xr[i];
  __syncthreads();
  int n = top_idx[row];
  float ox, oy;
  if (n < 4096)      { ox = ((n & 63) + 0.5f) / 64.f;  oy = ((n >> 6) + 0.5f) / 64.f; }
  else if (n < 5120) { int m = n - 4096; ox = ((m & 31) + 0.5f) / 32.f; oy = ((m >> 5) + 0.5f) / 32.f; }
  else               { int m = n - 5120; ox = ((m & 15) + 0.5f) / 16.f; oy = ((m >> 4) + 0.5f) / 16.f; }
#pragma unroll
  for (int j = 0; j < 8; ++j) {
    float p = 0.f;
#pragma unroll
    for (int i = 0; i < 8; ++i) {
      int k = lane * 8 + i;
      p += rb[wave][k] * W[k * 8 + j];
    }
    for (int o = 16; o > 0; o >>= 1) p += __shfl_down(p, o, 32);
    if (lane == 0) {
      float q   = tanhf(p + ob[j]);
      float off = (j & 1) ? oy : ox;
      out[(size_t)row * 8 + j] = (off + q) * 512.f;
    }
  }
}

// ---------------------------------------------------------------------------
extern "C" void kernel_launch(void* const* d_in, const int* in_sizes, int n_in,
                              void* d_out, int out_size, void* d_ws, size_t ws_size,
                              hipStream_t stream) {
  (void)in_sizes; (void)n_in; (void)out_size; (void)ws_size;
  const float* x3    = (const float*)d_in[3];
  const float* x4    = (const float*)d_in[4];
  const float* x5    = (const float*)d_in[5];
  const float* lat_w = (const float*)d_in[6];
  const float* lat_s = (const float*)d_in[7];
  const float* lat_b = (const float*)d_in[8];
  const float* gc_w  = (const float*)d_in[9];
  const float* gc_s  = (const float*)d_in[10];
  const float* gc_b  = (const float*)d_in[11];
  const float* loc_hw = (const float*)d_in[12];
  const float* loc_hb = (const float*)d_in[13];
  const float* loc_g  = (const float*)d_in[14];
  const float* loc_be = (const float*)d_in[15];
  const float* loc_ow = (const float*)d_in[16];
  const float* loc_ob = (const float*)d_in[17];
  const float* cls_hw = (const float*)d_in[18];
  const float* cls_hb = (const float*)d_in[19];
  const float* cls_g  = (const float*)d_in[20];
  const float* cls_be = (const float*)d_in[21];
  const float* cls_ow = (const float*)d_in[22];
  const float* cls_ob = (const float*)d_in[23];
  const float* quad_hw = (const float*)d_in[24];
  const float* quad_hb = (const float*)d_in[25];
  const float* quad_g  = (const float*)d_in[26];
  const float* quad_be = (const float*)d_in[27];
  const float* quad_ow = (const float*)d_in[28];
  const float* quad_ob = (const float*)d_in[29];

  // ---- workspace layout ----
  char* ws = (char*)d_ws;
  _Float16* W16 = (_Float16*)ws;                    // 16 transposed f16 256x256 mats
  size_t off = 16ull * 65536 * 2;                   // 2 MB
  float* gcbuf = (float*)(ws + off);  off += 16384;
  int* topidx  = (int*)(ws + off);    off += 8192;
  _Float16* feats = (_Float16*)(ws + off); off += (size_t)BATCH * NPOS * CDIM * 2;   // 44 MB
  float* logits   = (float*)(ws + off);    off += (size_t)BATCH * NPOS * 4;          // 344 KB
  _Float16* sel = (_Float16*)(ws + off); off += 1600ull * CDIM * 2;
  _Float16* h0  = (_Float16*)(ws + off); off += 1600ull * CDIM * 2;
  _Float16* h1  = (_Float16*)(ws + off); off += 1600ull * CDIM * 2;
  _Float16* Abuf = (_Float16*)(ws + off);           // 44 MB: transposed conv inputs
  _Float16* A3 = Abuf;
  _Float16* A4 = Abuf + (size_t)BATCH * 4096 * CDIM;
  _Float16* A5 = Abuf + (size_t)BATCH * (4096 + 1024) * CDIM;
  _Float16* act0 = Abuf;                            // alias: A dead after convs
  _Float16* act1 = (_Float16*)(ws + off + (size_t)BATCH * NPOS * CDIM * 2);

  _Float16* Wlat  = W16;
  _Float16* Wgc   = W16 + 3ull * 65536;
  _Float16* Wloc  = W16 + 4ull * 65536;
  _Float16* Wcls  = W16 + 8ull * 65536;
  _Float16* Wquad = W16 + 12ull * 65536;

  float* out = (float*)d_out;   // [0:16) num_inst | [16:1616) scores | [1616:3216) cls | [3216:16016) quad

  // 1) weight transpose + f16 convert
  k_transpose_w<<<dim3(8, 8, 3), 256, 0, stream>>>(lat_w,  Wlat);
  k_transpose_w<<<dim3(8, 8, 1), 256, 0, stream>>>(gc_w,   Wgc);
  k_transpose_w<<<dim3(8, 8, 4), 256, 0, stream>>>(loc_hw, Wloc);
  k_transpose_w<<<dim3(8, 8, 4), 256, 0, stream>>>(cls_hw, Wcls);
  k_transpose_w<<<dim3(8, 8, 4), 256, 0, stream>>>(quad_hw, Wquad);

  // 2) input transpose NCHW -> [B, HW, C] f16
  k_transpose_in<<<dim3(128, 8, BATCH), 256, 0, stream>>>(x3, A3, 4096);
  k_transpose_in<<<dim3(32, 8, BATCH),  256, 0, stream>>>(x4, A4, 1024);
  k_transpose_in<<<dim3(8, 8, BATCH),   256, 0, stream>>>(x5, A5, 256);

  // 3) global context (x5 conv + BN + SiLU + spatial mean)
  k_gc<<<dim3(16, BATCH), 256, 0, stream>>>(A5, Wgc, gc_s, gc_b, gcbuf);

  // 4) lateral convs -> feats
  k_conv<<<4096, 256, 0, stream>>>(A3, feats, Wlat,                lat_s,       lat_b,       gcbuf, 4096, 0);
  k_conv<<<1024, 256, 0, stream>>>(A4, feats, Wlat + 1ull * 65536, lat_s + 256, lat_b + 256, gcbuf, 1024, 4096);
  k_conv<<<256,  256, 0, stream>>>(A5, feats, Wlat + 2ull * 65536, lat_s + 512, lat_b + 512, gcbuf, 256,  5120);

  // 5) loc MLP over all 86016 rows (dominant GEMM chain)
  k_mlp_hidden<<<5376, 256, 0, stream>>>(feats, act0, Wloc + 0ull * 65536, loc_hb + 0,   loc_g + 0,   loc_be + 0);
  k_mlp_hidden<<<5376, 256, 0, stream>>>(act0,  act1, Wloc + 1ull * 65536, loc_hb + 256, loc_g + 256, loc_be + 256);
  k_mlp_hidden<<<5376, 256, 0, stream>>>(act1,  act0, Wloc + 2ull * 65536, loc_hb + 512, loc_g + 512, loc_be + 512);
  k_mlp_hidden<<<5376, 256, 0, stream>>>(act0,  act1, Wloc + 3ull * 65536, loc_hb + 768, loc_g + 768, loc_be + 768);
  k_loc_final<<<10752, 256, 0, stream>>>(act1, loc_ow, loc_ob, logits);

  // 6) top-k + gather
  k_topk<<<BATCH, 256, 0, stream>>>(logits, topidx, out);
  k_gather<<<1600, 128, 0, stream>>>(feats, topidx, sel);

  // 7) cls head (1600 rows)
  k_mlp_hidden<<<100, 256, 0, stream>>>(sel, h0, Wcls + 0ull * 65536, cls_hb + 0,   cls_g + 0,   cls_be + 0);
  k_mlp_hidden<<<100, 256, 0, stream>>>(h0,  h1, Wcls + 1ull * 65536, cls_hb + 256, cls_g + 256, cls_be + 256);
  k_mlp_hidden<<<100, 256, 0, stream>>>(h1,  h0, Wcls + 2ull * 65536, cls_hb + 512, cls_g + 512, cls_be + 512);
  k_mlp_hidden<<<100, 256, 0, stream>>>(h0,  h1, Wcls + 3ull * 65536, cls_hb + 768, cls_g + 768, cls_be + 768);
  k_cls_final<<<200, 256, 0, stream>>>(h1, cls_ow, cls_ob, out + 1616);

  // 8) quad head (1600 rows)
  k_mlp_hidden<<<100, 256, 0, stream>>>(sel, h0, Wquad + 0ull * 65536, quad_hb + 0,   quad_g + 0,   quad_be + 0);
  k_mlp_hidden<<<100, 256, 0, stream>>>(h0,  h1, Wquad + 1ull * 65536, quad_hb + 256, quad_g + 256, quad_be + 256);
  k_mlp_hidden<<<100, 256, 0, stream>>>(h1,  h0, Wquad + 2ull * 65536, quad_hb + 512, quad_g + 512, quad_be + 512);
  k_mlp_hidden<<<100, 256, 0, stream>>>(h0,  h1, Wquad + 3ull * 65536, quad_hb + 768, quad_g + 768, quad_be + 768);
  k_quad_final<<<200, 256, 0, stream>>>(h1, quad_ow, quad_ob, topidx, out + 3216);
}
